// ScaledDotProductAttention_5068061409509
// MI455X (gfx1250) — matrix operand order repaired and checked by reading, not compiled, not run on hardware
//
#include <hip/hip_runtime.h>
#include <hip/hip_bf16.h>

#define SEQ 4096
#define DIM 768
#define HID 512

typedef __attribute__((ext_vector_type(16))) __bf16 v16bf;
typedef __attribute__((ext_vector_type(8)))  __bf16 v8bf;
typedef __attribute__((ext_vector_type(8)))  float  v8f;

__device__ __forceinline__ v16bf pack16(v8bf lo, v8bf hi) {
  return __builtin_shufflevector(lo, hi, 0,1,2,3,4,5,6,7,8,9,10,11,12,13,14,15);
}

// ---------------------------------------------------------------------------
// Gate MLP: g[row] = tanh(v[row]@W1 + b1)@W2 + b2, folded into per-column
// affine coeffs a[j] = (1-g)/sqrt(768), c[j] = g/(S*sqrt(768)).
// 8 rows per block, thread t owns hidden units t and t+256.
// ---------------------------------------------------------------------------
__global__ void gate_kernel(const float* __restrict__ v, const float* __restrict__ W1,
                            const float* __restrict__ b1, const float* __restrict__ W2,
                            const float* __restrict__ b2,
                            float* __restrict__ acoef, float* __restrict__ bcoef) {
  __shared__ float vloc[8 * DIM];   // 24 KB
  __shared__ float red[8 * 256];    //  8 KB
  const int t = threadIdx.x;
  const int rowbase = blockIdx.x * 8;
  for (int i = t; i < 8 * DIM; i += 256) vloc[i] = v[(size_t)rowbase * DIM + i];
  __syncthreads();

  float acc[8][2];
#pragma unroll
  for (int r = 0; r < 8; ++r) { acc[r][0] = 0.f; acc[r][1] = 0.f; }

  for (int c = 0; c < DIM; ++c) {
    float w0 = W1[(size_t)c * HID + t];
    float w1 = W1[(size_t)c * HID + t + 256];
#pragma unroll
    for (int r = 0; r < 8; ++r) {
      float vc = vloc[r * DIM + c];
      acc[r][0] = fmaf(vc, w0, acc[r][0]);
      acc[r][1] = fmaf(vc, w1, acc[r][1]);
    }
  }
  const float w2a = W2[t], w2b = W2[t + 256];
  const float b1a = b1[t], b1b = b1[t + 256];
#pragma unroll
  for (int r = 0; r < 8; ++r) {
    float h0 = tanhf(acc[r][0] + b1a);
    float h1 = tanhf(acc[r][1] + b1b);
    red[r * 256 + t] = h0 * w2a + h1 * w2b;
  }
  __syncthreads();
  for (int off = 128; off > 0; off >>= 1) {
    if (t < off) {
#pragma unroll
      for (int r = 0; r < 8; ++r) red[r * 256 + t] += red[r * 256 + t + off];
    }
    __syncthreads();
  }
  if (t < 8) {
    const float inv = 0.036084391824351615f;  // 1/sqrt(768)
    float g = red[t * 256] + b2[0];
    acoef[rowbase + t] = (1.0f - g) * inv;
    bcoef[rowbase + t] = g * inv * (1.0f / (float)SEQ);  // m == 1/S exactly
  }
}

// ---------------------------------------------------------------------------
// fp32 -> bf16 elementwise convert (for q and k: contraction dim is contiguous
// in memory for both WMMA A (row-major) and B (column-major) operands).
// ---------------------------------------------------------------------------
__global__ void cvt_bf16_kernel(const float* __restrict__ src, __bf16* __restrict__ dst, int n) {
  for (int i = blockIdx.x * blockDim.x + threadIdx.x; i < n; i += gridDim.x * blockDim.x)
    dst[i] = (__bf16)src[i];
}

// ---------------------------------------------------------------------------
// v [SEQ,DIM] fp32 -> vtb [DIM,SEQ] bf16 (B operand of the second GEMM needs
// the contraction dim (rows of v) contiguous per lane => transpose once).
// ---------------------------------------------------------------------------
__global__ void transpose_bf16_kernel(const float* __restrict__ src, __bf16* __restrict__ dst) {
  __shared__ float tile[32][33];
  const int x = blockIdx.x * 32 + threadIdx.x;  // column (feature) 0..767
  const int y0 = blockIdx.y * 32;               // row base 0..4095
  for (int i = threadIdx.y; i < 32; i += 8)
    tile[i][threadIdx.x] = src[(size_t)(y0 + i) * DIM + x];
  __syncthreads();
  const int xo = y0 + threadIdx.x;              // dst column = src row
  const int yo = blockIdx.x * 32;               // dst row base = src column base
  for (int i = threadIdx.y; i < 32; i += 8)
    dst[(size_t)(yo + i) * SEQ + xo] = (__bf16)tile[threadIdx.x][i];
}

// ---------------------------------------------------------------------------
// attn_scores = (q @ k^T) * a[col] + c[col], bf16 WMMA, fp32 accumulate.
// Per wave: 64(M) x 32(N) tile (8 accs). Block: 8 waves (2x4) -> 128 x 128.
// A: lane L%16 = row, K split {0..7,16..23}/{8..15,24..31} by L/16.
// B (column-major per ISA 7.12.3): lane L%16 = col of k^T = row of k,
//   K split {0..7,16..23}/{8..15,24..31} by L/16 (same as A). No transpose needed.
// ---------------------------------------------------------------------------
__global__ void scores_kernel(const __bf16* __restrict__ qb, const __bf16* __restrict__ kb16,
                              const float* __restrict__ acoef, const float* __restrict__ bcoef,
                              float* __restrict__ Wout) {
  const int lane = threadIdx.x & 31;
  const int wave = threadIdx.x >> 5;
  const int wm = wave & 1, wn = wave >> 1;           // 2 x 4 wave grid
  const int mbase = blockIdx.y * 128 + wm * 64;
  const int nbase = blockIdx.x * 128 + wn * 32;
  const int r = lane & 15, hi = lane >> 4;

  v8f acc[4][2];
#pragma unroll
  for (int i = 0; i < 4; ++i)
#pragma unroll
    for (int j = 0; j < 2; ++j)
#pragma unroll
      for (int e = 0; e < 8; ++e) acc[i][j][e] = 0.f;

  for (int kb = 0; kb < DIM; kb += 32) {
    v16bf afrag[4], bfrag[2];
#pragma unroll
    for (int mt = 0; mt < 4; ++mt) {
      const __bf16* ap = qb + (size_t)(mbase + mt * 16 + r) * DIM + kb + hi * 8;
      afrag[mt] = pack16(*(const v8bf*)ap, *(const v8bf*)(ap + 16));
    }
#pragma unroll
    for (int nt = 0; nt < 2; ++nt) {
      const __bf16* bp = kb16 + (size_t)(nbase + nt * 16 + r) * DIM + kb + hi * 8;
      bfrag[nt] = pack16(*(const v8bf*)bp, *(const v8bf*)(bp + 16));
    }
#pragma unroll
    for (int mt = 0; mt < 4; ++mt)
#pragma unroll
      for (int nt = 0; nt < 2; ++nt)
        acc[mt][nt] = __builtin_amdgcn_wmma_f32_16x16x32_bf16(
            false, afrag[mt], false, bfrag[nt], (short)0, acc[mt][nt], false, false);
  }

  // C/D layout: lane column = r, VGPR e -> row e + 8*hi. Per-column affine.
#pragma unroll
  for (int mt = 0; mt < 4; ++mt)
#pragma unroll
    for (int nt = 0; nt < 2; ++nt) {
      const int col = nbase + nt * 16 + r;
      const float a = acoef[col], c = bcoef[col];
      const int row0 = mbase + mt * 16 + hi * 8;
#pragma unroll
      for (int e = 0; e < 8; ++e)
        Wout[(size_t)(row0 + e) * SEQ + col] = fmaf(acc[mt][nt][e], a, c);
    }
}

// ---------------------------------------------------------------------------
// In-place numerically-stable row softmax over the [SEQ,SEQ] weights region.
// One block per row; row staged in LDS (16 KB).
// ---------------------------------------------------------------------------
__global__ void softmax_kernel(float* __restrict__ W) {
  __shared__ float buf[SEQ];
  __shared__ float red[256];
  const int t = threadIdx.x;
  float* row = W + (size_t)blockIdx.x * SEQ;

  float m = -3.402823466e38f;
  for (int i = t; i < SEQ; i += 256) { float x = row[i]; buf[i] = x; m = fmaxf(m, x); }
  red[t] = m; __syncthreads();
  for (int off = 128; off > 0; off >>= 1) { if (t < off) red[t] = fmaxf(red[t], red[t + off]); __syncthreads(); }
  m = red[0]; __syncthreads();

  float s = 0.f;
  for (int i = t; i < SEQ; i += 256) { float e = __expf(buf[i] - m); buf[i] = e; s += e; }
  red[t] = s; __syncthreads();
  for (int off = 128; off > 0; off >>= 1) { if (t < off) red[t] += red[t + off]; __syncthreads(); }
  const float invs = 1.0f / red[0];

  for (int i = t; i < SEQ; i += 256) row[i] = buf[i] * invs;
}

// ---------------------------------------------------------------------------
// attn_output = attn_weights @ v.  A = weights (fp32 -> bf16 in registers),
// B = vtb (v^T, bf16, contraction contiguous). Same 64x32-per-wave tiling.
// ---------------------------------------------------------------------------
__global__ void out_kernel(const float* __restrict__ Wn, const __bf16* __restrict__ vtb,
                           float* __restrict__ out) {
  const int lane = threadIdx.x & 31;
  const int wave = threadIdx.x >> 5;
  const int wm = wave & 1, wn = wave >> 1;           // 2 x 4 wave grid
  const int mbase = blockIdx.y * 128 + wm * 64;
  const int nbase = blockIdx.x * 128 + wn * 32;
  const int r = lane & 15, hi = lane >> 4;

  v8f acc[4][2];
#pragma unroll
  for (int i = 0; i < 4; ++i)
#pragma unroll
    for (int j = 0; j < 2; ++j)
#pragma unroll
      for (int e = 0; e < 8; ++e) acc[i][j][e] = 0.f;

  for (int kb = 0; kb < SEQ; kb += 32) {
    v16bf afrag[4], bfrag[2];
#pragma unroll
    for (int mt = 0; mt < 4; ++mt) {
      const float* ap = Wn + (size_t)(mbase + mt * 16 + r) * SEQ + kb + hi * 8;
      float4 f0 = *(const float4*)ap;
      float4 f1 = *(const float4*)(ap + 4);
      float4 f2 = *(const float4*)(ap + 16);
      float4 f3 = *(const float4*)(ap + 20);
      v16bf a;
      a[0]  = (__bf16)f0.x; a[1]  = (__bf16)f0.y; a[2]  = (__bf16)f0.z; a[3]  = (__bf16)f0.w;
      a[4]  = (__bf16)f1.x; a[5]  = (__bf16)f1.y; a[6]  = (__bf16)f1.z; a[7]  = (__bf16)f1.w;
      a[8]  = (__bf16)f2.x; a[9]  = (__bf16)f2.y; a[10] = (__bf16)f2.z; a[11] = (__bf16)f2.w;
      a[12] = (__bf16)f3.x; a[13] = (__bf16)f3.y; a[14] = (__bf16)f3.z; a[15] = (__bf16)f3.w;
      afrag[mt] = a;
    }
#pragma unroll
    for (int nt = 0; nt < 2; ++nt) {
      const __bf16* bp = vtb + (size_t)(nbase + nt * 16 + r) * SEQ + kb + hi * 8;
      bfrag[nt] = pack16(*(const v8bf*)bp, *(const v8bf*)(bp + 16));
    }
#pragma unroll
    for (int mt = 0; mt < 4; ++mt)
#pragma unroll
      for (int nt = 0; nt < 2; ++nt)
        acc[mt][nt] = __builtin_amdgcn_wmma_f32_16x16x32_bf16(
            false, afrag[mt], false, bfrag[nt], (short)0, acc[mt][nt], false, false);
  }

#pragma unroll
  for (int mt = 0; mt < 4; ++mt)
#pragma unroll
    for (int nt = 0; nt < 2; ++nt) {
      const int col = nbase + nt * 16 + r;
      const int row0 = mbase + mt * 16 + hi * 8;
#pragma unroll
      for (int e = 0; e < 8; ++e)
        out[(size_t)(row0 + e) * DIM + col] = acc[mt][nt][e];
    }
}

// ---------------------------------------------------------------------------
extern "C" void kernel_launch(void* const* d_in, const int* in_sizes, int n_in,
                              void* d_out, int out_size, void* d_ws, size_t ws_size,
                              hipStream_t stream) {
  (void)in_sizes; (void)n_in; (void)out_size; (void)ws_size;
  const float* q  = (const float*)d_in[0];
  const float* k  = (const float*)d_in[1];
  const float* v  = (const float*)d_in[2];
  const float* W1 = (const float*)d_in[3];
  const float* b1 = (const float*)d_in[4];
  const float* W2 = (const float*)d_in[5];
  const float* b2 = (const float*)d_in[6];

  float* out_attn = (float*)d_out;                       // [SEQ, DIM]
  float* Wout     = out_attn + (size_t)SEQ * DIM;        // [SEQ, SEQ] weights

  __bf16* qb   = (__bf16*)d_ws;                          // [SEQ, DIM]
  __bf16* kb16 = qb   + (size_t)SEQ * DIM;               // [SEQ, DIM]
  __bf16* vtb  = kb16 + (size_t)SEQ * DIM;               // [DIM, SEQ] (v^T)
  float*  acoef = (float*)(vtb + (size_t)DIM * SEQ);     // [SEQ]
  float*  bcoef = acoef + SEQ;                           // [SEQ]

  gate_kernel<<<SEQ / 8, 256, 0, stream>>>(v, W1, b1, W2, b2, acoef, bcoef);
  cvt_bf16_kernel<<<2048, 256, 0, stream>>>(q, qb, SEQ * DIM);
  cvt_bf16_kernel<<<2048, 256, 0, stream>>>(k, kb16, SEQ * DIM);
  transpose_bf16_kernel<<<dim3(DIM / 32, SEQ / 32), dim3(32, 8), 0, stream>>>(v, vtb);
  scores_kernel<<<dim3(SEQ / 128, SEQ / 128), 256, 0, stream>>>(qb, kb16, acoef, bcoef, Wout);
  softmax_kernel<<<SEQ, 256, 0, stream>>>(Wout);
  out_kernel<<<dim3(DIM / 128, SEQ / 128), 256, 0, stream>>>(Wout, vtb, out_attn);
}
